// ObjectEncoder_33715493273843
// MI455X (gfx1250) — matrix operand order repaired and checked by reading, not compiled, not run on hardware
//
#include <hip/hip_runtime.h>

typedef _Float16 h8  __attribute__((ext_vector_type(8)));
typedef _Float16 h16 __attribute__((ext_vector_type(16)));
typedef float    f8  __attribute__((ext_vector_type(8)));

#define EPSBN 1e-5f

// ---------------------------------------------------------------------------
// Utility kernels
// ---------------------------------------------------------------------------
__global__ void k_zero(float* p, int n) {
  int i = blockIdx.x * blockDim.x + threadIdx.x;
  if (i < n) p[i] = 0.f;
}

// Convert f32 weights [Cin,Cout] -> f16 transposed+padded [Cout, Kp]
__global__ void k_cvt_w(const float* __restrict__ W, _Float16* __restrict__ Wt,
                        int Cin, int Cout, int Kp) {
  int i = blockIdx.x * blockDim.x + threadIdx.x;
  if (i >= Cout * Kp) return;
  int k = i % Kp, co = i / Kp;
  Wt[i] = (k < Cin) ? (_Float16)W[(size_t)k * Cout + co] : (_Float16)0.f;
}

// f32 -> f16 (optionally ReLU)
__global__ void k_cvt(const float* __restrict__ s, _Float16* __restrict__ d,
                      size_t n, int relu) {
  size_t i = (size_t)blockIdx.x * blockDim.x + threadIdx.x;
  size_t stride = (size_t)gridDim.x * blockDim.x;
  for (; i < n; i += stride) {
    float v = s[i];
    if (relu) v = fmaxf(v, 0.f);
    d[i] = (_Float16)v;
  }
}

// ---------------------------------------------------------------------------
// Farthest point sampling: one block per cloud, N<=1024, 256 threads
// ---------------------------------------------------------------------------
__global__ void k_fps(const float* __restrict__ pos, int* __restrict__ out,
                      int N, int m) {
  __shared__ float sx[1024], sy[1024], sz[1024];
  __shared__ float rv[256];
  __shared__ int   ri[256];
  __shared__ int   s_last;
  const float* p = pos + (size_t)blockIdx.x * N * 3;
  int* o = out + (size_t)blockIdx.x * m;
  int tid = threadIdx.x;
  for (int j = tid; j < N; j += 256) {
    sx[j] = p[3 * j + 0]; sy[j] = p[3 * j + 1]; sz[j] = p[3 * j + 2];
  }
  float mind[4];
  #pragma unroll
  for (int q = 0; q < 4; ++q) mind[q] = 3.4e38f;
  if (tid == 0) { o[0] = 0; s_last = 0; }
  __syncthreads();
  for (int it = 1; it < m; ++it) {
    int last = s_last;
    float lx = sx[last], ly = sy[last], lz = sz[last];
    float bv = -1.f; int bi = 0x7fffffff;
    int q = 0;
    for (int j = tid; j < N; j += 256, ++q) {
      float dx = sx[j] - lx, dy = sy[j] - ly, dz = sz[j] - lz;
      float d = dx * dx + dy * dy + dz * dz;
      float mn = fminf(mind[q], d);
      mind[q] = mn;
      if (mn > bv || (mn == bv && j < bi)) { bv = mn; bi = j; }
    }
    rv[tid] = bv; ri[tid] = bi;
    __syncthreads();
    for (int s = 128; s > 0; s >>= 1) {
      if (tid < s) {
        float ov = rv[tid + s]; int oi = ri[tid + s];
        if (ov > rv[tid] || (ov == rv[tid] && oi < ri[tid])) {
          rv[tid] = ov; ri[tid] = oi;
        }
      }
      __syncthreads();
    }
    if (tid == 0) { o[it] = ri[0]; s_last = ri[0]; }
    __syncthreads();
  }
}

// gather 3-vectors: out[b,i,:] = pos[b, idx[b,i], :]
__global__ void k_gather3(const float* __restrict__ pos, const int* __restrict__ idx,
                          float* __restrict__ out, int B, int N, int M) {
  int g = blockIdx.x * blockDim.x + threadIdx.x;
  if (g >= B * M) return;
  int b = g / M;
  int j = idx[g];
  const float* s = pos + ((size_t)b * N + j) * 3;
  float* d = out + (size_t)g * 3;
  d[0] = s[0]; d[1] = s[1]; d[2] = s[2];
}

// radius neighbors: first K (=64) in-radius points by index (matches -idx top_k)
__global__ void k_radius(const float* __restrict__ pos, const float* __restrict__ posq,
                         int* __restrict__ nbr, float* __restrict__ mask,
                         int B, int N, int M, float r2) {
  int g = blockIdx.x * blockDim.x + threadIdx.x;
  if (g >= B * M) return;
  int b = g / M;
  const float* p = pos + (size_t)b * N * 3;
  float qx = posq[(size_t)g * 3 + 0];
  float qy = posq[(size_t)g * 3 + 1];
  float qz = posq[(size_t)g * 3 + 2];
  int* nb = nbr + (size_t)g * 64;
  float* mk = mask + (size_t)g * 64;
  int cnt = 0;
  for (int j = 0; j < N && cnt < 64; ++j) {
    float dx = p[3 * j + 0] - qx, dy = p[3 * j + 1] - qy, dz = p[3 * j + 2] - qz;
    float d2 = dx * dx + dy * dy + dz * dz;
    if (d2 <= r2) { nb[cnt] = j; mk[cnt] = 1.f; ++cnt; }
  }
  for (; cnt < 64; ++cnt) { nb[cnt] = 0; mk[cnt] = 0.f; }
}

// SA1 edge features: rel pos, f16 padded to 32 channels
__global__ void k_edges1(const float* __restrict__ pos, const float* __restrict__ pos1,
                         const int* __restrict__ nbr, _Float16* __restrict__ A,
                         int B, int N, int M) {
  int e = blockIdx.x * blockDim.x + threadIdx.x;
  if (e >= B * M * 64) return;
  int g = e >> 6;          // b*M + i
  int b = g / M;
  int j = nbr[e];
  const float* pj = pos + ((size_t)b * N + j) * 3;
  const float* q = pos1 + (size_t)g * 3;
  _Float16* a = A + (size_t)e * 32;
  a[0] = (_Float16)(pj[0] - q[0]);
  a[1] = (_Float16)(pj[1] - q[1]);
  a[2] = (_Float16)(pj[2] - q[2]);
  #pragma unroll
  for (int c = 3; c < 32; ++c) a[c] = (_Float16)0.f;
}

// SA2 edge features: [x1[nbr] (128), rel pos (3)] -> f16 padded to 160
__global__ void k_edges2(const float* __restrict__ pos1, const float* __restrict__ pos2,
                         const int* __restrict__ nbr, const float* __restrict__ x1,
                         _Float16* __restrict__ A, int B, int M1, int M2) {
  int e = blockIdx.x * blockDim.x + threadIdx.x;
  if (e >= B * M2 * 64) return;
  int g = e >> 6;
  int b = g / M2;
  int j = nbr[e];
  const float* xf = x1 + ((size_t)b * M1 + j) * 128;
  const float* pj = pos1 + ((size_t)b * M1 + j) * 3;
  const float* q = pos2 + (size_t)g * 3;
  _Float16* a = A + (size_t)e * 160;
  for (int c = 0; c < 128; ++c) a[c] = (_Float16)xf[c];
  a[128] = (_Float16)(pj[0] - q[0]);
  a[129] = (_Float16)(pj[1] - q[1]);
  a[130] = (_Float16)(pj[2] - q[2]);
  #pragma unroll
  for (int c = 131; c < 160; ++c) a[c] = (_Float16)0.f;
}

// global module input: [x2 (256), pos2 (3)] -> f16 padded to 288
__global__ void k_concat_g(const float* __restrict__ x2, const float* __restrict__ pos2,
                           _Float16* __restrict__ A, int BM) {
  int g = blockIdx.x * blockDim.x + threadIdx.x;
  if (g >= BM) return;
  const float* x = x2 + (size_t)g * 256;
  _Float16* a = A + (size_t)g * 288;
  for (int c = 0; c < 256; ++c) a[c] = (_Float16)x[c];
  a[256] = (_Float16)pos2[(size_t)g * 3 + 0];
  a[257] = (_Float16)pos2[(size_t)g * 3 + 1];
  a[258] = (_Float16)pos2[(size_t)g * 3 + 2];
  #pragma unroll
  for (int c = 259; c < 288; ++c) a[c] = (_Float16)0.f;
}

// ---------------------------------------------------------------------------
// GEMM: Y[E,Cout] = A[E,Kp](f16) @ Wt[Cout,Kp]^T(f16) + bias, via WMMA f16
// One wave per 16x16 tile, 4 waves per block. E, Cout multiples of 16; Kp of 32.
// ---------------------------------------------------------------------------
__global__ void __launch_bounds__(128)
k_gemm(const _Float16* __restrict__ A, const _Float16* __restrict__ Wt,
       const float* __restrict__ bias, float* __restrict__ Y,
       int E, int Kp, int Cout) {
  int wave = threadIdx.x >> 5;
  int lane = threadIdx.x & 31;
  int tilesM = E >> 4, tilesN = Cout >> 4;
  int t = blockIdx.x * 4 + wave;
  if (t >= tilesM * tilesN) return;
  int tn = t / tilesM, tm = t % tilesM;
  int l16 = lane & 15;
  int arow = (tm << 4) + l16;          // A-fragment row
  int col = (tn << 4) + l16;           // B-fragment / output column
  int ka = (lane < 16) ? 0 : 8;        // A K-group split per ISA layout
  int kb = (lane < 16) ? 0 : 8;        // B K split per ISA layout
  const _Float16* ap = A + (size_t)arow * Kp;
  const _Float16* bp = Wt + (size_t)col * Kp;
  f8 acc = {0.f, 0.f, 0.f, 0.f, 0.f, 0.f, 0.f, 0.f};
  for (int k0 = 0; k0 < Kp; k0 += 32) {
    union { h16 v; h8 p[2]; } af, bf;
    af.p[0] = *(const h8*)(ap + k0 + ka);
    af.p[1] = *(const h8*)(ap + k0 + ka + 16);
    bf.p[0] = *(const h8*)(bp + k0 + kb);
    bf.p[1] = *(const h8*)(bp + k0 + kb + 16);
    acc = __builtin_amdgcn_wmma_f32_16x16x32_f16(false, af.v, false, bf.v,
                                                 (short)0, acc, false, false);
  }
  float bb = bias ? bias[col] : 0.f;
  int r0 = (lane < 16) ? 0 : 8;        // D layout: vgpr r -> row r (+8 for hi lanes)
  #pragma unroll
  for (int r = 0; r < 8; ++r)
    Y[(size_t)((tm << 4) + r0 + r) * Cout + col] = acc[r] + bb;
}

// ---------------------------------------------------------------------------
// BatchNorm (training stats, optional mask) : stats = [sum(C), sumsq(C), count]
// ---------------------------------------------------------------------------
__global__ void k_bn_stats(const float* __restrict__ Y, const float* __restrict__ mask,
                           float* __restrict__ stats, int E, int C) {
  int tid = threadIdx.x;  // 256 threads; each owns channels tid, tid+256, ...
  float s[4] = {0.f, 0.f, 0.f, 0.f};
  float q[4] = {0.f, 0.f, 0.f, 0.f};
  float cnt = 0.f;
  for (int r = blockIdx.x; r < E; r += gridDim.x) {
    if (mask && mask[r] == 0.f) continue;
    const float* y = Y + (size_t)r * C;
    #pragma unroll
    for (int u = 0; u < 4; ++u) {
      int c = tid + u * 256;
      if (c < C) { float v = y[c]; s[u] += v; q[u] += v * v; }
    }
    if (tid == 0) cnt += 1.f;
  }
  #pragma unroll
  for (int u = 0; u < 4; ++u) {
    int c = tid + u * 256;
    if (c < C) {
      atomicAdd(&stats[c], s[u]);
      atomicAdd(&stats[C + c], q[u]);
    }
  }
  if (tid == 0 && cnt != 0.f) atomicAdd(&stats[2 * C], cnt);
}

// normalize + scale/shift + ReLU -> f16 (next layer's A matrix)
__global__ void k_bn_apply(const float* __restrict__ Y, const float* __restrict__ stats,
                           const float* __restrict__ gam, const float* __restrict__ bet,
                           _Float16* __restrict__ H, size_t total, int C) {
  size_t i = (size_t)blockIdx.x * blockDim.x + threadIdx.x;
  size_t stride = (size_t)gridDim.x * blockDim.x;
  float n = fmaxf(stats[2 * C], 1.f);
  for (; i < total; i += stride) {
    int c = (int)(i % C);
    float mean = stats[c] / n;
    float var = stats[C + c] / n - mean * mean;
    float v = gam[c] * (Y[i] - mean) * rsqrtf(var + EPSBN) + bet[c];
    H[i] = (_Float16)fmaxf(v, 0.f);
  }
}

// masked max over the K axis: X[r,c] = max_k (mask[r,k] ? Y[r,k,c] : -inf)
__global__ void k_max(const float* __restrict__ Y, const float* __restrict__ mask,
                      float* __restrict__ X, int BM, int Kn, int C) {
  int g = blockIdx.x * blockDim.x + threadIdx.x;
  if (g >= BM * C) return;
  int c = g % C, r = g / C;
  const float* y = Y + (size_t)r * Kn * C + c;
  float best = -3.4e38f;
  for (int k = 0; k < Kn; ++k) {
    if (!mask || mask[(size_t)r * Kn + k] != 0.f)
      best = fmaxf(best, y[(size_t)k * C]);
  }
  X[g] = best;
}

// ---------------------------------------------------------------------------
// Host launch
// ---------------------------------------------------------------------------
extern "C" void kernel_launch(void* const* d_in, const int* in_sizes, int n_in,
                              void* d_out, int out_size, void* d_ws, size_t ws_size,
                              hipStream_t stream) {
  (void)in_sizes; (void)n_in; (void)out_size; (void)ws_size;
  const int B = 16, N = 1024, M1 = 512, M2 = 128, K = 64;
  const int E1 = B * M1 * K;   // 524288
  const int E2 = B * M2 * K;   // 131072
  const int E3 = B * M2;       // 2048

  auto F = [&](int i) { return (const float*)d_in[i]; };
  const float* pos = F(0);

  // bump allocator over ws
  char* ws = (char*)d_ws;
  size_t off = 0;
  auto alloc = [&](size_t bytes) -> void* {
    off = (off + 255) & ~(size_t)255;
    void* p = ws + off;
    off += bytes;
    return p;
  };

  // transposed f16 weights: {input idx, Cin, Cout, Kpad}
  struct WI { int idx, cin, cout, kp; };
  const WI wl[12] = {
      {1, 3, 64, 32},     {2, 64, 64, 64},     {3, 64, 128, 64},
      {11, 131, 128, 160},{12, 128, 128, 128}, {13, 128, 256, 128},
      {21, 259, 256, 288},{22, 256, 512, 256}, {23, 512, 1024, 512},
      {31, 1024, 512, 1024},{32, 512, 256, 512},{33, 256, 32, 256}};
  _Float16* wt[12];
  for (int i = 0; i < 12; ++i)
    wt[i] = (_Float16*)alloc((size_t)wl[i].cout * wl[i].kp * sizeof(_Float16));

  int* idx1   = (int*)alloc((size_t)B * M1 * sizeof(int));
  float* pos1 = (float*)alloc((size_t)B * M1 * 3 * sizeof(float));
  int* idx2   = (int*)alloc((size_t)B * M2 * sizeof(int));
  float* pos2 = (float*)alloc((size_t)B * M2 * 3 * sizeof(float));
  int* nbr1   = (int*)alloc((size_t)E1 * sizeof(int));
  float* msk1 = (float*)alloc((size_t)E1 * sizeof(float));
  int* nbr2   = (int*)alloc((size_t)E2 * sizeof(int));
  float* msk2 = (float*)alloc((size_t)E2 * sizeof(float));
  float* x1   = (float*)alloc((size_t)B * M1 * 128 * sizeof(float));
  float* x2   = (float*)alloc((size_t)B * M2 * 256 * sizeof(float));
  float* gbuf = (float*)alloc((size_t)B * 1024 * sizeof(float));
  float* stats = (float*)alloc((2 * 1024 + 1) * sizeof(float));
  _Float16* hbuf = (_Float16*)alloc((size_t)E1 * 64 * sizeof(_Float16)); // max A
  float* ybuf = (float*)alloc((size_t)E1 * 128 * sizeof(float));         // max Y

  // weight conversion
  for (int i = 0; i < 12; ++i) {
    int n = wl[i].cout * wl[i].kp;
    k_cvt_w<<<(n + 255) / 256, 256, 0, stream>>>(F(wl[i].idx), wt[i],
                                                 wl[i].cin, wl[i].cout, wl[i].kp);
  }

  auto gemm = [&](const _Float16* Aq, const _Float16* Wq, const float* bq,
                  float* Yq, int E, int Kp, int Cout) {
    int tiles = (E >> 4) * (Cout >> 4);
    k_gemm<<<(tiles + 3) / 4, 128, 0, stream>>>(Aq, Wq, bq, Yq, E, Kp, Cout);
  };
  auto bnrelu = [&](const float* Yq, const float* mk, const float* gg,
                    const float* bb, _Float16* Hq, int E, int C) {
    k_zero<<<(2 * C + 1 + 255) / 256, 256, 0, stream>>>(stats, 2 * C + 1);
    int sb = E < 512 ? E : 512;
    k_bn_stats<<<sb, 256, 0, stream>>>(Yq, mk, stats, E, C);
    size_t tot = (size_t)E * C;
    int ab = (int)((tot + 255) / 256); if (ab > 16384) ab = 16384;
    k_bn_apply<<<ab, 256, 0, stream>>>(Yq, stats, gg, bb, Hq, tot, C);
  };

  // -------------------- SA1 --------------------
  k_fps<<<B, 256, 0, stream>>>(pos, idx1, N, M1);
  k_gather3<<<(B * M1 + 127) / 128, 128, 0, stream>>>(pos, idx1, pos1, B, N, M1);
  k_radius<<<(B * M1 + 127) / 128, 128, 0, stream>>>(pos, pos1, nbr1, msk1,
                                                     B, N, M1, 0.04f);
  k_edges1<<<(E1 + 255) / 256, 256, 0, stream>>>(pos, pos1, nbr1, hbuf, B, N, M1);

  gemm(hbuf, wt[0], F(4), ybuf, E1, 32, 64);
  bnrelu(ybuf, msk1, F(7), F(9), hbuf, E1, 64);
  gemm(hbuf, wt[1], F(5), ybuf, E1, 64, 64);
  bnrelu(ybuf, msk1, F(8), F(10), hbuf, E1, 64);
  gemm(hbuf, wt[2], F(6), ybuf, E1, 64, 128);
  k_max<<<(B * M1 * 128 + 255) / 256, 256, 0, stream>>>(ybuf, msk1, x1,
                                                        B * M1, K, 128);

  // -------------------- SA2 --------------------
  k_fps<<<B, 256, 0, stream>>>(pos1, idx2, M1, M2);
  k_gather3<<<(B * M2 + 127) / 128, 128, 0, stream>>>(pos1, idx2, pos2, B, M1, M2);
  k_radius<<<(B * M2 + 127) / 128, 128, 0, stream>>>(pos1, pos2, nbr2, msk2,
                                                     B, M1, M2, 0.16f);
  k_edges2<<<(E2 + 255) / 256, 256, 0, stream>>>(pos1, pos2, nbr2, x1, hbuf,
                                                 B, M1, M2);

  gemm(hbuf, wt[3], F(14), ybuf, E2, 160, 128);
  bnrelu(ybuf, msk2, F(17), F(19), hbuf, E2, 128);
  gemm(hbuf, wt[4], F(15), ybuf, E2, 128, 128);
  bnrelu(ybuf, msk2, F(18), F(20), hbuf, E2, 128);
  gemm(hbuf, wt[5], F(16), ybuf, E2, 128, 256);
  k_max<<<(B * M2 * 256 + 255) / 256, 256, 0, stream>>>(ybuf, msk2, x2,
                                                        B * M2, K, 256);

  // -------------------- Global SA --------------------
  k_concat_g<<<(E3 + 255) / 256, 256, 0, stream>>>(x2, pos2, hbuf, E3);
  gemm(hbuf, wt[6], F(24), ybuf, E3, 288, 256);
  bnrelu(ybuf, (const float*)nullptr, F(27), F(29), hbuf, E3, 256);
  gemm(hbuf, wt[7], F(25), ybuf, E3, 256, 512);
  bnrelu(ybuf, (const float*)nullptr, F(28), F(30), hbuf, E3, 512);
  gemm(hbuf, wt[8], F(26), ybuf, E3, 512, 1024);
  k_max<<<(B * 1024 + 255) / 256, 256, 0, stream>>>(ybuf, (const float*)nullptr,
                                                    gbuf, B, M2, 1024);

  // -------------------- Final plain MLP --------------------
  k_cvt<<<64, 256, 0, stream>>>(gbuf, hbuf, (size_t)B * 1024, 0);
  gemm(hbuf, wt[9], F(34), ybuf, B, 1024, 512);
  k_cvt<<<32, 256, 0, stream>>>(ybuf, hbuf, (size_t)B * 512, 1);
  gemm(hbuf, wt[10], F(35), ybuf, B, 512, 256);
  k_cvt<<<16, 256, 0, stream>>>(ybuf, hbuf, (size_t)B * 256, 1);
  gemm(hbuf, wt[11], F(36), (float*)d_out, B, 256, 32);
}
